// Attention_mca_g2l_12103217840309
// MI455X (gfx1250) — matrix operand order repaired and checked by reading, not compiled, not run on hardware
//
#include <hip/hip_runtime.h>
#include <hip/hip_bf16.h>

typedef __bf16 bf16_t;
typedef __bf16 v16bf __attribute__((ext_vector_type(16)));
typedef float  v8f   __attribute__((ext_vector_type(8)));

#define N1   512
#define N2   2048
#define CDIM 1024
#define C2   2048
#define NH   8
#define HD   128
#define OUTW 3072   /* 3*C */

// gfx1250 async global->LDS path, feature-guarded so compile can't regress.
#if defined(__gfx1250__) && \
    __has_builtin(__builtin_amdgcn_global_load_async_to_lds_b128) && \
    __has_builtin(__builtin_amdgcn_s_wait_asynccnt)
#define HAVE_ASYNC_LDS 1
#else
#define HAVE_ASYNC_LDS 0
#endif

#define AS1 __attribute__((address_space(1)))
#define AS3 __attribute__((address_space(3)))
typedef int vsi4 __attribute__((vector_size(4 * sizeof(int))));

#if HAVE_ASYNC_LDS
#define ASYNC_CP16(gp, lp)                                              \
  __builtin_amdgcn_global_load_async_to_lds_b128(                       \
      (AS1 vsi4*)(unsigned long long)(uintptr_t)(gp),                   \
      (AS3 vsi4*)(unsigned long long)(uintptr_t)(lp), 0, 0)
#else
#define ASYNC_CP16(gp, lp) (*(uint4*)(lp) = *(const uint4*)(gp))
#endif

__device__ __forceinline__ bf16_t f2bf(float f)  { return (bf16_t)f; }
__device__ __forceinline__ float  bf2f(bf16_t b) { return (float)b; }

// ---------------------------------------------------------------- conversions
__global__ void k_cvt(bf16_t* __restrict__ dst, const float* __restrict__ src, int n) {
  int i = blockIdx.x * blockDim.x + threadIdx.x;
  if (i < n) dst[i] = f2bf(src[i]);
}

__global__ void k_cvt_strided(bf16_t* __restrict__ dst, int dld,
                              const float* __restrict__ src, int sld, int soff,
                              int rows, int cols) {
  int i = blockIdx.x * blockDim.x + threadIdx.x;
  if (i >= rows * cols) return;
  int r = i / cols, c = i % cols;
  dst[(size_t)r * dld + c] = f2bf(src[(size_t)r * sld + soff + c]);
}

// per-(row,head) L2 normalization of 128-wide head slices, output bf16
__global__ void k_l2norm(bf16_t* __restrict__ dst, int dld,
                         const float* __restrict__ src, int sld, int soff, int rows) {
  int i = blockIdx.x * blockDim.x + threadIdx.x;
  if (i >= rows * NH) return;
  int r = i / NH, h = i % NH;
  const float* s = src + (size_t)r * sld + soff + h * HD;
  float ss = 0.0f;
  for (int d = 0; d < HD; ++d) { float v = s[d]; ss += v * v; }
  float inv = rsqrtf(ss);
  bf16_t* o = dst + (size_t)r * dld + h * HD;
  for (int d = 0; d < HD; ++d) o[d] = f2bf(s[d] * inv);
}

// ---------------------------------------------------------------- WMMA GEMM
// C[M,N] = A[M,K] * op(B) (+ bias), bf16 in / f32 out.
// TRANSB=false: B stored [K,N] row-major.  TRANSB=true: B stored [N,K] -> A*B^T.
// M % 128 == 0, N % 128 == 0, K % 64 == 0 (all call sites satisfy this).
// 8 waves in a 4(M) x 2(N) grid; each wave computes a 32x64 tile (2x4 WMMA accs)
// -> 8 WMMAs per 6 fragment loads per 32-K step.
#define BM   128
#define BN   128
#define BKK  64
#define APAD 8
#define BPAD 8

union Frag { uint4 q[2]; v16bf v; };

template <bool TRANSB>
__global__ __launch_bounds__(256) void k_gemm(
    const bf16_t* __restrict__ A, int lda,
    const bf16_t* __restrict__ B, int ldb,
    float* __restrict__ C, int ldc,
    int M, int N, int K, const float* __restrict__ bias) {
  (void)M; (void)N;
  __shared__ alignas(16) bf16_t As[BM][BKK + APAD];
  // B tile stored TRANSPOSED in LDS: BsT[n][k] -> each lane's B fragment is two
  // 16B k-runs 32B apart (two ds_load_b128), same as the A fragment.
  __shared__ alignas(16) bf16_t BsT[BN][BKK + BPAD];

  const int tid  = threadIdx.x;
  const int lane = tid & 31;
  const int wave = tid >> 5;            // 8 waves
  const int wm   = (wave >> 1) * 32;    // 0/32/64/96
  const int wn   = (wave & 1)  * 64;    // 0/64
  const int bm0  = blockIdx.y * BM;
  const int bn0  = blockIdx.x * BN;
  const int l15  = lane & 15;
  const int lhi  = lane >> 4;

  v8f acc[2][4];
  #pragma unroll
  for (int i = 0; i < 2; i++)
    #pragma unroll
    for (int j = 0; j < 4; j++)
      #pragma unroll
      for (int e = 0; e < 8; e++) acc[i][j][e] = 0.0f;

  for (int k0 = 0; k0 < K; k0 += BKK) {
    if (k0 + BKK < K) {   // gfx1250 global_prefetch of next A tile
      const bf16_t* pf = A + (size_t)(bm0 + (tid >> 1)) * lda + (k0 + BKK) + (tid & 1) * 32;
      __builtin_prefetch(pf, 0, 0);
    }
    __syncthreads();
    // ---- stage A: 128x64 bf16, 16B chunks, 4 per thread (async DMA to LDS)
    #pragma unroll
    for (int t = 0; t < 4; ++t) {
      int vi = tid + t * 256;
      int r  = vi >> 3;            // 0..127
      int c  = (vi & 7) << 3;      // 0..56
      ASYNC_CP16(A + (size_t)(bm0 + r) * lda + k0 + c, &As[r][c]);
    }
    // ---- stage B (transposed into LDS)
    if (TRANSB) {
      // global [N][K]: contiguous on both sides, 16B chunks (async DMA to LDS)
      #pragma unroll
      for (int t = 0; t < 4; ++t) {
        int vi = tid + t * 256;
        int n  = vi >> 3;            // 0..127
        int c  = (vi & 7) << 3;      // 0..56
        ASYNC_CP16(B + (size_t)(bn0 + n) * ldb + k0 + c, &BsT[n][c]);
      }
    } else {
      // global [K][N]: vector read along n, transpose-scatter into LDS
      #pragma unroll
      for (int t = 0; t < 8; ++t) {
        int vi = tid + t * 256;
        int k  = vi >> 5;            // 0..63
        int n4 = (vi & 31) << 2;     // 0..124
        const bf16_t* g = B + (size_t)(k0 + k) * ldb + bn0 + n4;
        bf16_t e0 = g[0], e1 = g[1], e2 = g[2], e3 = g[3];
        BsT[n4 + 0][k] = e0; BsT[n4 + 1][k] = e1;
        BsT[n4 + 2][k] = e2; BsT[n4 + 3][k] = e3;
      }
    }
#if HAVE_ASYNC_LDS
    __builtin_amdgcn_s_wait_asynccnt(0);
#endif
    __syncthreads();

    #pragma unroll
    for (int kk = 0; kk < BKK; kk += 32) {
      v16bf af[2], bfr[4];
      // A fragment: lane holds K = lhi*8 + {0..7} and 16 + lhi*8 + {0..7}
      #pragma unroll
      for (int i = 0; i < 2; i++) {
        int mr = wm + i * 16 + l15;
        Frag fa;
        fa.q[0] = *(const uint4*)&As[mr][kk + lhi * 8];
        fa.q[1] = *(const uint4*)&As[mr][kk + 16 + lhi * 8];
        af[i] = fa.v;
      }
      // B fragment: lane holds K = lhi*8 + {0..7} and 16 + lhi*8 + {0..7} at column nc
      #pragma unroll
      for (int j = 0; j < 4; j++) {
        int nc = wn + j * 16 + l15;
        const bf16_t* pb = &BsT[nc][kk + lhi * 8];
        Frag fb;
        fb.q[0] = *(const uint4*)(pb);
        fb.q[1] = *(const uint4*)(pb + 16);
        bfr[j] = fb.v;
      }
      #pragma unroll
      for (int i = 0; i < 2; i++)
        #pragma unroll
        for (int j = 0; j < 4; j++)
          acc[i][j] = __builtin_amdgcn_wmma_f32_16x16x32_bf16(
              false, af[i], false, bfr[j], (short)0, acc[i][j], false, false);
    }
  }

  // epilogue: f32 C/D layout (VGPR v -> M = v + 8*lhi, N = lane&15)
  #pragma unroll
  for (int i = 0; i < 2; i++) {
    #pragma unroll
    for (int j = 0; j < 4; j++) {
      int row0 = bm0 + wm + i * 16 + lhi * 8;
      int col  = bn0 + wn + j * 16 + l15;
      float badd = bias ? bias[col] : 0.0f;
      #pragma unroll
      for (int v = 0; v < 8; ++v)
        C[(size_t)(row0 + v) * ldc + col] = acc[i][j][v] + badd;
    }
  }
}

// ---------------------------------------------------------------- reductions
__device__ __forceinline__ float blk_max(float v, float* sm) {
  int t = threadIdx.x;
  sm[t] = v; __syncthreads();
  for (int s = 128; s > 0; s >>= 1) { if (t < s) sm[t] = fmaxf(sm[t], sm[t + s]); __syncthreads(); }
  float r = sm[0]; __syncthreads(); return r;
}
__device__ __forceinline__ float blk_sum(float v, float* sm) {
  int t = threadIdx.x;
  sm[t] = v; __syncthreads();
  for (int s = 128; s > 0; s >>= 1) { if (t < s) sm[t] += sm[t + s]; __syncthreads(); }
  float r = sm[0]; __syncthreads(); return r;
}

// blended softmax for one head: attn = 0.5*(softmax(Sc*25*cs) + softmax(Sr*25))
__global__ __launch_bounds__(256) void k_softmax_avg(
    const float* __restrict__ Sc, const float* __restrict__ Sr,
    const float* __restrict__ cs, bf16_t* __restrict__ attn_h) {
  __shared__ float sm[256];
  int q = blockIdx.x, t = threadIdx.x;
  const float* rc = Sc + (size_t)q * N2;
  const float* rr = Sr + (size_t)q * N2;
  float lc[8], lr[8];
  float mc = -1e30f, mr = -1e30f;
  #pragma unroll
  for (int e = 0; e < 8; e++) {
    int k = t + e * 256;
    lc[e] = rc[k] * 25.0f * cs[k];
    lr[e] = rr[k] * 25.0f;
    mc = fmaxf(mc, lc[e]); mr = fmaxf(mr, lr[e]);
  }
  mc = blk_max(mc, sm); mr = blk_max(mr, sm);
  float sc = 0.0f, sr = 0.0f;
  #pragma unroll
  for (int e = 0; e < 8; e++) {
    lc[e] = __expf(lc[e] - mc); sc += lc[e];
    lr[e] = __expf(lr[e] - mr); sr += lr[e];
  }
  sc = blk_sum(sc, sm); sr = blk_sum(sr, sm);
  float ic = 1.0f / sc, ir = 1.0f / sr;
  #pragma unroll
  for (int e = 0; e < 8; e++)
    attn_h[(size_t)q * N2 + t + e * 256] = f2bf(0.5f * (lc[e] * ic + lr[e] * ir));
}

__global__ void k_head_sum(const bf16_t* __restrict__ attn, float* __restrict__ hsum) {
  int i = blockIdx.x * blockDim.x + threadIdx.x;
  if (i >= N1 * N2) return;
  float s = 0.0f;
  #pragma unroll
  for (int h = 0; h < NH; ++h) s += bf2f(attn[(size_t)h * N1 * N2 + i]);
  hsum[i] = s * (1.0f / NH);
}

// second-round softmax + similarity/objectness masking + renormalization
__global__ __launch_bounds__(256) void k_round2(
    const float* __restrict__ hsum, const float* __restrict__ rawc,
    const float* __restrict__ rawr, bf16_t* __restrict__ s_out,
    bf16_t* __restrict__ o_out) {
  __shared__ float sm[256];
  int q = blockIdx.x, t = threadIdx.x;
  const float* hr = hsum + (size_t)q * N2;
  float l[8]; float m = -1e30f;
  #pragma unroll
  for (int e = 0; e < 8; e++) { l[e] = hr[t + e * 256]; m = fmaxf(m, l[e]); }
  m = blk_max(m, sm);
  float s = 0.0f;
  #pragma unroll
  for (int e = 0; e < 8; e++) { l[e] = __expf(l[e] - m); s += l[e]; }
  s = blk_sum(s, sm);
  float inv = 1.0f / s;
  float sv[8]; float ssum = 0.0f;
  #pragma unroll
  for (int e = 0; e < 8; e++) {
    int k = t + e * 256;
    float mask = (rawc[(size_t)q * N2 + k] > 6.0f) ? 1.0f : 0.0f;   // 0.75*H
    sv[e] = mask * l[e] * inv; ssum += sv[e];
  }
  ssum = blk_sum(ssum, sm);
  float is2 = 1.0f / ssum;
  float ov[8]; float osum = 0.0f;
  #pragma unroll
  for (int e = 0; e < 8; e++) {
    int k = t + e * 256;
    sv[e] *= is2;
    float om = (rawr[(size_t)q * N2 + k] > 7.92f) ? 1.0f : 0.0f;    // 0.99*H
    ov[e] = om * sv[e]; osum += ov[e];
  }
  osum = blk_sum(osum, sm);
  float io = 1.0f / osum;
  #pragma unroll
  for (int e = 0; e < 8; e++) {
    int k = t + e * 256;
    s_out[(size_t)q * N2 + k] = f2bf(sv[e]);
    o_out[(size_t)q * N2 + k] = f2bf(ov[e] * io);
  }
}

// build [x_att | v_local] bf16 concat for the output linear
__global__ void k_concat(bf16_t* __restrict__ dst, const float* __restrict__ xatt,
                         const bf16_t* __restrict__ vb) {
  int i = blockIdx.x * blockDim.x + threadIdx.x;
  if (i >= N1 * C2) return;
  int r = i / C2, c = i % C2;
  dst[i] = (c < CDIM) ? f2bf(xatt[(size_t)r * CDIM + c])
                      : vb[(size_t)r * CDIM + (c - CDIM)];
}

__global__ void k_copy_out(float* __restrict__ dst, const float* __restrict__ src) {
  int i = blockIdx.x * blockDim.x + threadIdx.x;
  if (i >= N1 * C2) return;
  int r = i / C2, c = i % C2;
  dst[(size_t)r * OUTW + CDIM + c] = src[i];
}

// ---------------------------------------------------------------- launcher
extern "C" void kernel_launch(void* const* d_in, const int* in_sizes, int n_in,
                              void* d_out, int out_size, void* d_ws, size_t ws_size,
                              hipStream_t stream) {
  (void)in_sizes; (void)n_in; (void)out_size; (void)ws_size;
  const float* x_cls     = (const float*)d_in[0];
  const float* x_reg     = (const float*)d_in[1];
  const float* cls_score = (const float*)d_in[2];
  const float* W_q_cls   = (const float*)d_in[4];
  const float* W_kv_cls  = (const float*)d_in[5];
  const float* W_q_reg   = (const float*)d_in[6];
  const float* W_kv_reg  = (const float*)d_in[7];
  const float* W_lin     = (const float*)d_in[8];
  const float* b_lin     = (const float*)d_in[9];
  const float* W_lin_reg = (const float*)d_in[10];
  const float* b_lin_reg = (const float*)d_in[11];
  float* out = (float*)d_out;

  char* p = (char*)d_ws;
  auto alloc = [&](size_t bytes) { void* r = (void*)p; p += (bytes + 255) & ~(size_t)255; return r; };

  bf16_t* xcls_b  = (bf16_t*)alloc((size_t)N2 * CDIM * 2);
  bf16_t* xreg_b  = (bf16_t*)alloc((size_t)N2 * CDIM * 2);
  bf16_t* wqc_b   = (bf16_t*)alloc((size_t)CDIM * CDIM * 2);
  bf16_t* wkvc_b  = (bf16_t*)alloc((size_t)CDIM * C2 * 2);
  bf16_t* wqr_b   = (bf16_t*)alloc((size_t)CDIM * CDIM * 2);
  bf16_t* wkvr_b  = (bf16_t*)alloc((size_t)CDIM * C2 * 2);
  bf16_t* wlin_b  = (bf16_t*)alloc((size_t)C2 * C2 * 2);
  bf16_t* wlinr_b = (bf16_t*)alloc((size_t)C2 * C2 * 2);
  float*  qcls_f  = (float*)alloc((size_t)N1 * CDIM * 4);
  float*  qreg_f  = (float*)alloc((size_t)N1 * CDIM * 4);
  float*  kvc_f   = (float*)alloc((size_t)N2 * C2 * 4);
  float*  kvr_f   = (float*)alloc((size_t)N2 * C2 * 4);
  bf16_t* qnc_b   = (bf16_t*)alloc((size_t)N1 * CDIM * 2);
  bf16_t* qnr_b   = (bf16_t*)alloc((size_t)N1 * CDIM * 2);
  bf16_t* knc_b   = (bf16_t*)alloc((size_t)N2 * CDIM * 2);
  bf16_t* knr_b   = (bf16_t*)alloc((size_t)N2 * CDIM * 2);
  bf16_t* vnc_b   = (bf16_t*)alloc((size_t)N2 * CDIM * 2);
  bf16_t* vnr_b   = (bf16_t*)alloc((size_t)N2 * CDIM * 2);
  bf16_t* vc_b    = (bf16_t*)alloc((size_t)N2 * CDIM * 2);
  bf16_t* vr_b    = (bf16_t*)alloc((size_t)N2 * CDIM * 2);
  float*  raws_c  = (float*)alloc((size_t)N1 * N2 * 4);
  float*  raws_r  = (float*)alloc((size_t)N1 * N2 * 4);
  float*  Sc      = (float*)alloc((size_t)N1 * N2 * 4);
  float*  Sr      = (float*)alloc((size_t)N1 * N2 * 4);
  bf16_t* attn_b  = (bf16_t*)alloc((size_t)NH * N1 * N2 * 2);
  float*  hsum    = (float*)alloc((size_t)N1 * N2 * 4);
  float*  xatt_c  = (float*)alloc((size_t)N1 * CDIM * 4);
  float*  xatt_r  = (float*)alloc((size_t)N1 * CDIM * 4);
  bf16_t* cat_c   = (bf16_t*)alloc((size_t)N1 * C2 * 2);
  bf16_t* cat_r   = (bf16_t*)alloc((size_t)N1 * C2 * 2);
  float*  outc    = (float*)alloc((size_t)N1 * C2 * 4);
  float*  outr    = (float*)alloc((size_t)N1 * C2 * 4);
  bf16_t* s_b     = (bf16_t*)alloc((size_t)N1 * N2 * 2);
  bf16_t* o_b     = (bf16_t*)alloc((size_t)N1 * N2 * 2);

  auto g1 = [](int n) { return dim3((n + 255) / 256); };

  // 1) bf16 conversions of inputs and weights
  k_cvt<<<g1(N2 * CDIM), 256, 0, stream>>>(xcls_b, x_cls, N2 * CDIM);
  k_cvt<<<g1(N2 * CDIM), 256, 0, stream>>>(xreg_b, x_reg, N2 * CDIM);
  k_cvt<<<g1(CDIM * CDIM), 256, 0, stream>>>(wqc_b, W_q_cls, CDIM * CDIM);
  k_cvt<<<g1(CDIM * C2), 256, 0, stream>>>(wkvc_b, W_kv_cls, CDIM * C2);
  k_cvt<<<g1(CDIM * CDIM), 256, 0, stream>>>(wqr_b, W_q_reg, CDIM * CDIM);
  k_cvt<<<g1(CDIM * C2), 256, 0, stream>>>(wkvr_b, W_kv_reg, CDIM * C2);
  k_cvt<<<g1(C2 * C2), 256, 0, stream>>>(wlin_b, W_lin, C2 * C2);
  k_cvt<<<g1(C2 * C2), 256, 0, stream>>>(wlinr_b, W_lin_reg, C2 * C2);

  // 2) projections
  k_gemm<false><<<dim3(CDIM / BN, N1 / BM), 256, 0, stream>>>(xcls_b, CDIM, wqc_b, CDIM, qcls_f, CDIM, N1, CDIM, CDIM, nullptr);
  k_gemm<false><<<dim3(CDIM / BN, N1 / BM), 256, 0, stream>>>(xreg_b, CDIM, wqr_b, CDIM, qreg_f, CDIM, N1, CDIM, CDIM, nullptr);
  k_gemm<false><<<dim3(C2 / BN, N2 / BM), 256, 0, stream>>>(xcls_b, CDIM, wkvc_b, C2, kvc_f, C2, N2, C2, CDIM, nullptr);
  k_gemm<false><<<dim3(C2 / BN, N2 / BM), 256, 0, stream>>>(xreg_b, CDIM, wkvr_b, C2, kvr_f, C2, N2, C2, CDIM, nullptr);

  // 3) per-head L2 normalization + v extraction (k = cols [0,C), v = cols [C,2C))
  k_l2norm<<<g1(N1 * NH), 256, 0, stream>>>(qnc_b, CDIM, qcls_f, CDIM, 0, N1);
  k_l2norm<<<g1(N1 * NH), 256, 0, stream>>>(qnr_b, CDIM, qreg_f, CDIM, 0, N1);
  k_l2norm<<<g1(N2 * NH), 256, 0, stream>>>(knc_b, CDIM, kvc_f, C2, 0, N2);
  k_l2norm<<<g1(N2 * NH), 256, 0, stream>>>(knr_b, CDIM, kvr_f, C2, 0, N2);
  k_l2norm<<<g1(N2 * NH), 256, 0, stream>>>(vnc_b, CDIM, kvc_f, C2, CDIM, N2);
  k_l2norm<<<g1(N2 * NH), 256, 0, stream>>>(vnr_b, CDIM, kvr_f, C2, CDIM, N2);
  k_cvt_strided<<<g1(N2 * CDIM), 256, 0, stream>>>(vc_b, CDIM, kvc_f, C2, CDIM, N2, CDIM);
  k_cvt_strided<<<g1(N2 * CDIM), 256, 0, stream>>>(vr_b, CDIM, kvr_f, C2, CDIM, N2, CDIM);

  // 4) head-summed raw v-v similarity collapses to one full-C GEMM per branch
  k_gemm<true><<<dim3(N2 / BN, N1 / BM), 256, 0, stream>>>(vnc_b, CDIM, vnc_b, CDIM, raws_c, N2, N1, N2, CDIM, nullptr);
  k_gemm<true><<<dim3(N2 / BN, N1 / BM), 256, 0, stream>>>(vnr_b, CDIM, vnr_b, CDIM, raws_r, N2, N1, N2, CDIM, nullptr);

  // 5) per-head attention: logits -> blended softmax -> attn x V
  for (int h = 0; h < NH; ++h) {
    const bf16_t* qc = qnc_b + h * HD;
    const bf16_t* kc = knc_b + h * HD;
    const bf16_t* qr = qnr_b + h * HD;
    const bf16_t* kr = knr_b + h * HD;
    bf16_t* ah = attn_b + (size_t)h * N1 * N2;
    k_gemm<true><<<dim3(N2 / BN, N1 / BM), 256, 0, stream>>>(qc, CDIM, kc, CDIM, Sc, N2, N1, N2, HD, nullptr);
    k_gemm<true><<<dim3(N2 / BN, N1 / BM), 256, 0, stream>>>(qr, CDIM, kr, CDIM, Sr, N2, N1, N2, HD, nullptr);
    k_softmax_avg<<<dim3(N1), 256, 0, stream>>>(Sc, Sr, cls_score, ah);
    k_gemm<false><<<dim3(HD / BN, N1 / BM), 256, 0, stream>>>(ah, N2, vc_b + h * HD, CDIM, xatt_c + h * HD, CDIM, N1, HD, N2, nullptr);
    k_gemm<false><<<dim3(HD / BN, N1 / BM), 256, 0, stream>>>(ah, N2, vr_b + h * HD, CDIM, xatt_r + h * HD, CDIM, N1, HD, N2, nullptr);
  }

  // 6) second-round weights
  k_head_sum<<<g1(N1 * N2), 256, 0, stream>>>(attn_b, hsum);
  k_round2<<<dim3(N1), 256, 0, stream>>>(hsum, raws_c, raws_r, s_b, o_b);

  // 7) output linears on [x_att | v_local]
  k_concat<<<g1(N1 * C2), 256, 0, stream>>>(cat_c, xatt_c, vc_b);
  k_concat<<<g1(N1 * C2), 256, 0, stream>>>(cat_r, xatt_r, vr_b);
  k_gemm<false><<<dim3(C2 / BN, N1 / BM), 256, 0, stream>>>(cat_c, C2, wlin_b, C2, outc, C2, N1, C2, C2, b_lin);
  k_gemm<false><<<dim3(C2 / BN, N1 / BM), 256, 0, stream>>>(cat_r, C2, wlinr_b, C2, outr, C2, N1, C2, C2, b_lin_reg);

  // 8) final features: [sim_round2 @ support | out] written straight into d_out
  k_gemm<false><<<dim3(CDIM / BN, N1 / BM), 256, 0, stream>>>(s_b, N2, vc_b, CDIM, out, OUTW, N1, CDIM, N2, nullptr);
  k_gemm<false><<<dim3(CDIM / BN, N1 / BM), 256, 0, stream>>>(o_b, N2, vr_b, CDIM, out + (size_t)N1 * OUTW, OUTW, N1, CDIM, N2, nullptr);
  k_copy_out<<<g1(N1 * C2), 256, 0, stream>>>(out, outc);
  k_copy_out<<<g1(N1 * C2), 256, 0, stream>>>(out + (size_t)N1 * OUTW, outr);
}
